// GNNGaussianencoder_12652973654572
// MI455X (gfx1250) — matrix operand order repaired and checked by reading, not compiled, not run on hardware
//
#include <hip/hip_runtime.h>
#include <hip/hip_bf16.h>
#include <math.h>

typedef __attribute__((ext_vector_type(16))) __bf16 v16bf;
typedef __attribute__((ext_vector_type(8)))  __bf16 v8bf;
typedef __attribute__((ext_vector_type(4)))  __bf16 v4bf;
typedef __attribute__((ext_vector_type(8)))  float  v8f;

#define DD 128
#define HH 256
#define KK 50
#define BB 1000
#define EPSBN 1e-5f

#define WMMA_BF16(A, B, C) \
  __builtin_amdgcn_wmma_f32_16x16x32_bf16(false, (A), false, (B), (short)0, (C), false, false)

// ---------------- WMMA fragment loaders (16x16x32 bf16, contiguous k = 16h + s in A and B) -------
// A (16x32, row-major): lane l -> row m0+(l&15); half h=l>>4 reads bytes
// [32h..32h+31] of the 64B row span -> two aligned 16B loads.
__device__ __forceinline__ v16bf frag_a_bf16(const __bf16* __restrict__ src, int lda,
                                             int rows, int m0, int k0) {
  int l = threadIdx.x & 31;
  int m = m0 + (l & 15); if (m >= rows) m = rows - 1;
  int h = l >> 4;
  const __bf16* p = src + (size_t)m * lda + k0 + 16 * h;
  v8bf lo = *(const v8bf*)p;
  v8bf hi = *(const v8bf*)(p + 8);
  v16bf a;
#pragma unroll
  for (int v = 0; v < 8; ++v) { a[v] = lo[v]; a[8 + v] = hi[v]; }
  return a;
}

// B (32x16) from row-major bf16 weight W[out,in] in global memory.
__device__ __forceinline__ v16bf frag_b_bf16(const __bf16* __restrict__ W, int ldw,
                                             int nrows, int n0, int k0) {
  int l = threadIdx.x & 31;
  int n = n0 + (l & 15); if (n >= nrows) n = nrows - 1;
  int h = l >> 4;
  const __bf16* p = W + (size_t)n * ldw + k0 + 16 * h;
  v8bf lo = *(const v8bf*)p;
  v8bf hi = *(const v8bf*)(p + 8);
  v16bf b;
#pragma unroll
  for (int v = 0; v < 8; ++v) { b[v] = lo[v]; b[8 + v] = hi[v]; }
  return b;
}

// B fragment from an LDS-staged weight panel (padded row stride ldp, local
// row nloc): 16B ds loads, bank-conflict-free with ldp = cols + 8.
__device__ __forceinline__ v16bf frag_b_lds(const __bf16* S, int ldp, int nloc, int k0) {
  int l = threadIdx.x & 31;
  const __bf16* p = S + (size_t)(nloc + (l & 15)) * ldp + k0 + 16 * (l >> 4);
  v8bf lo = *(const v8bf*)p;
  v8bf hi = *(const v8bf*)(p + 8);
  v16bf b;
#pragma unroll
  for (int v = 0; v < 8; ++v) { b[v] = lo[v]; b[8 + v] = hi[v]; }
  return b;
}

__device__ __forceinline__ void copy16(__bf16* dst, const __bf16* src) {
  *(v8bf*)dst       = *(const v8bf*)src;
  *(v8bf*)(dst + 8) = *(const v8bf*)(src + 8);
}

// ---------------- utility kernels ------------------------------------------
__global__ void k_zero(float* p, size_t count) {
  size_t i = (size_t)blockIdx.x * blockDim.x + threadIdx.x;
  size_t st = (size_t)gridDim.x * blockDim.x;
  for (; i < count; i += st) p[i] = 0.f;
}

__global__ void k_cast_bf16(const float* __restrict__ src, __bf16* __restrict__ dst,
                            size_t count) {
  size_t i = (size_t)blockIdx.x * blockDim.x + threadIdx.x;
  size_t st = (size_t)gridDim.x * blockDim.x;
  for (; i < count; i += st) dst[i] = (__bf16)src[i];
}

__global__ void k_gather(const int* __restrict__ idx, const float* __restrict__ wv,
                         __bf16* __restrict__ xbf, int total) {
  int i = blockIdx.x * blockDim.x + threadIdx.x;
  if (i >= total) return;
  int row = i >> 7, c = i & (DD - 1);
  xbf[i] = (__bf16)wv[(size_t)idx[row] * DD + c];
}

// scatter-add over edges (+ appended self-loops): 1 wave/edge, 4 cols/lane
__global__ void k_edges(const int* __restrict__ eidx, const float* __restrict__ ew,
                        const float* __restrict__ idx_w, const __bf16* __restrict__ xbf,
                        float* __restrict__ aggr, int E, int total) {
  int e = blockIdx.x * 8 + (threadIdx.x >> 5);
  if (e >= total) return;
  int lane = threadIdx.x & 31;
  int src, dst; float w;
  if (e < E) { src = eidx[e]; dst = eidx[E + e]; w = ew[e]; }
  else       { src = dst = e - E; w = idx_w[e - E]; }
  v4bf xs = *(const v4bf*)(xbf + (size_t)src * DD + lane * 4);
  float* ad = aggr + (size_t)dst * DD + lane * 4;
#pragma unroll
  for (int j = 0; j < 4; ++j) atomicAdd(&ad[j], w * (float)xs[j]);
}

// gnn = aggr @ w_rel^T + b_rel + x @ w_root^T.  A panels in registers (read
// once); weights staged cooperatively into LDS in two 64-column halves.
__global__ void k_gemm_gnn(const __bf16* __restrict__ aggrbf, const __bf16* __restrict__ xbf,
                           const __bf16* __restrict__ w_relbf, const __bf16* __restrict__ w_rootbf,
                           const float* __restrict__ b_rel, float* __restrict__ gnn, int n) {
  constexpr int LDP = DD + 8;                      // padded LDS row stride
  __shared__ __bf16 sB[2][64 * LDP];               // ~34 KB
  int wave = threadIdx.x >> 5;
  int mt = blockIdx.x * 8 + wave;
  bool valid = mt * 16 < n;
  int m0 = valid ? mt * 16 : 0;
  bool full = m0 + 16 <= n;
  v16bf a1[4], a2[4];
#pragma unroll
  for (int i = 0; i < 4; ++i) {
    a1[i] = frag_a_bf16(aggrbf, DD, n, m0, i * 32);
    a2[i] = frag_a_bf16(xbf,    DD, n, m0, i * 32);
  }
  int l = threadIdx.x & 31, h = l >> 4;
  for (int half = 0; half < 2; ++half) {
    int nbase = half * 64;
    __syncthreads();
    for (int t = threadIdx.x; t < 64 * 8; t += 256) {   // 8 x 16-elem chunks per row
      int row = t >> 3, ck = (t & 7) * 16;
      copy16(&sB[0][row * LDP + ck], w_relbf  + (size_t)(nbase + row) * DD + ck);
      copy16(&sB[1][row * LDP + ck], w_rootbf + (size_t)(nbase + row) * DD + ck);
    }
    __syncthreads();
    for (int nt = 0; nt < 4; ++nt) {
      int nl = nt * 16;
      v8f acc = {};
#pragma unroll
      for (int i = 0; i < 4; ++i) {
        acc = WMMA_BF16(a1[i], frag_b_lds(sB[0], LDP, nl, i * 32), acc);
        acc = WMMA_BF16(a2[i], frag_b_lds(sB[1], LDP, nl, i * 32), acc);
      }
      if (!valid) continue;
      int col = nbase + nl + (l & 15);
      float bias = b_rel[col];
      float* g = gnn + (size_t)(m0 + 8 * h) * DD + col;
      if (full) {
#pragma unroll
        for (int r = 0; r < 8; ++r) g[(size_t)r * DD] = acc[r] + bias;
      } else {
#pragma unroll
        for (int r = 0; r < 8; ++r)
          if (m0 + r + 8 * h < n) g[(size_t)r * DD] = acc[r] + bias;
      }
    }
  }
}

// column sums / sumsq (training-mode batchnorm statistics)
__global__ void k_colstats(const float* __restrict__ src, int rows, int cols, int ld,
                           int rows_per_block, float* __restrict__ sums,
                           float* __restrict__ sumsq) {
  int r0 = blockIdx.x * rows_per_block;
  int rend = r0 + rows_per_block; if (rend > rows) rend = rows;
  for (int c = threadIdx.x; c < cols; c += blockDim.x) {
    float s = 0.f, s2 = 0.f;
    for (int r = r0 + threadIdx.y; r < rend; r += blockDim.y) {
      float v = src[(size_t)r * ld + c];
      s += v; s2 += v * v;
    }
    atomicAdd(&sums[c], s);
    atomicAdd(&sumsq[c], s2);
  }
}

__global__ void k_finalize_stats(const float* __restrict__ sums, const float* __restrict__ sumsq,
                                 int cols, float inv_n, float* __restrict__ mean,
                                 float* __restrict__ rstd) {
  int c = blockIdx.x * blockDim.x + threadIdx.x;
  if (c >= cols) return;
  float m = sums[c] * inv_n;
  float v = sumsq[c] * inv_n - m * m;
  mean[c] = m;
  rstd[c] = rsqrtf(v + EPSBN);
}

// enc1_bf[i] = [ tanh(BN(gnn[i])) , x_bf[i] ]  (bf16, [N, 256])
__global__ void k_enc1(const float* __restrict__ gnn, const __bf16* __restrict__ xbf,
                       const float* __restrict__ mean, const float* __restrict__ rstd,
                       const float* __restrict__ g, const float* __restrict__ b,
                       __bf16* __restrict__ enc1bf, int total) {
  int i = blockIdx.x * blockDim.x + threadIdx.x;
  if (i >= total) return;
  int row = i >> 7, c = i & (DD - 1);
  float v = tanhf(g[c] * (gnn[i] - mean[c]) * rstd[c] + b[c]);
  enc1bf[(size_t)row * (2 * DD) + c] = (__bf16)v;
  enc1bf[(size_t)row * (2 * DD) + DD + c] = xbf[i];
}

// enc2 += segment_sum( sigmoid(enc1@w1^T+b1) * tanh(enc1@w2^T+b2) ).
// A panel (full K=256 row-tile) in registers; both weight matrices staged
// through LDS in 32-column stages; gated product + f32 atomics into [B,H].
__global__ void k_fc_scatter(const __bf16* __restrict__ enc1bf, const __bf16* __restrict__ w1bf,
                             const float* __restrict__ bi1, const __bf16* __restrict__ w2bf,
                             const float* __restrict__ bi2, const int* __restrict__ xbatch,
                             float* __restrict__ enc2, int n) {
  constexpr int LDP = 2 * DD + 8;                  // 264
  __shared__ __bf16 sW[2][32 * LDP];               // ~33 KB
  int wave = threadIdx.x >> 5;
  int mt = blockIdx.x * 8 + wave;
  bool valid = mt * 16 < n;
  int m0 = valid ? mt * 16 : 0;
  bool full = m0 + 16 <= n;
  v16bf a[8];
#pragma unroll
  for (int i = 0; i < 8; ++i) a[i] = frag_a_bf16(enc1bf, 2 * DD, n, m0, i * 32);
  int l = threadIdx.x & 31, h = l >> 4;
  for (int stage = 0; stage < 8; ++stage) {
    int nbase = stage * 32;
    __syncthreads();
    for (int t = threadIdx.x; t < 32 * 16; t += 256) {  // 16 x 16-elem chunks per row
      int row = t >> 4, ck = (t & 15) * 16;
      copy16(&sW[0][row * LDP + ck], w1bf + (size_t)(nbase + row) * (2 * DD) + ck);
      copy16(&sW[1][row * LDP + ck], w2bf + (size_t)(nbase + row) * (2 * DD) + ck);
    }
    __syncthreads();
    for (int nt = 0; nt < 2; ++nt) {
      int nl = nt * 16;
      v8f acc1 = {}, acc2 = {};
#pragma unroll
      for (int i = 0; i < 8; ++i) {
        acc1 = WMMA_BF16(a[i], frag_b_lds(sW[0], LDP, nl, i * 32), acc1);
        acc2 = WMMA_BF16(a[i], frag_b_lds(sW[1], LDP, nl, i * 32), acc2);
      }
      if (!valid) continue;
      int col = nbase + nl + (l & 15);
      float c1 = bi1[col], c2 = bi2[col];
#pragma unroll
      for (int r = 0; r < 8; ++r) {
        int m = m0 + r + 8 * h;
        if (full || m < n) {
          float u = acc1[r] + c1, v = acc2[r] + c2;
          float gate = 1.f / (1.f + __expf(-u));
          atomicAdd(&enc2[(size_t)xbatch[m] * HH + col], gate * tanhf(v));
        }
      }
    }
  }
}

// small GEMM: out[M,ncols] = A_bf16[M,kdim] @ W_bf16[ncols,kdim]^T (+bias)
__global__ void k_gemm_small(const __bf16* __restrict__ A, int lda, int M,
                             const __bf16* __restrict__ W, int ldw, int nrows,
                             const float* __restrict__ bias, float* __restrict__ out,
                             int ldo, int ncols, int kdim) {
  int m0 = blockIdx.x * 16;
  int n0 = blockIdx.y * 16;
  v8f acc = {};
  for (int k0 = 0; k0 < kdim; k0 += 32) {
    v16bf a = frag_a_bf16(A, lda, M, m0, k0);
    v16bf b = frag_b_bf16(W, ldw, nrows, n0, k0);
    acc = WMMA_BF16(a, b, acc);
  }
  int l = threadIdx.x & 31, col = n0 + (l & 15), h = l >> 4;
  if (col >= ncols) return;
  float bv = bias ? bias[col] : 0.f;
#pragma unroll
  for (int r = 0; r < 8; ++r) {
    int m = m0 + r + 8 * h;
    if (m < M) out[(size_t)m * ldo + col] = acc[r] + bv;
  }
}

// logits[i,k] = enc1[i]@w_phi[:, :2D]^T + doc_phi[x_batch[i],k] + b_phi[k]
// Token-part weight (50 rows x 256) staged once into LDS (rows padded/clamped
// to 64 so fragment reads need no guards).
__global__ void k_phi_logits(const __bf16* __restrict__ enc1bf, const __bf16* __restrict__ wphibf,
                             const float* __restrict__ b_phi, const float* __restrict__ docphi,
                             const int* __restrict__ xbatch, float* __restrict__ logits, int n) {
  constexpr int LDP = 2 * DD + 8;                  // 264
  __shared__ __bf16 sP[64 * LDP];                  // ~33 KB
  for (int t = threadIdx.x; t < 64 * 16; t += 256) {
    int row = t >> 4, ck = (t & 15) * 16;
    int gr = row < KK ? row : KK - 1;
    copy16(&sP[row * LDP + ck], wphibf + (size_t)gr * (2 * DD + HH) + ck);
  }
  __syncthreads();
  int wave = threadIdx.x >> 5;
  int mt = blockIdx.x * 8 + wave;
  bool valid = mt * 16 < n;
  int m0 = valid ? mt * 16 : 0;
  bool full = m0 + 16 <= n;
  v16bf a[8];
#pragma unroll
  for (int i = 0; i < 8; ++i) a[i] = frag_a_bf16(enc1bf, 2 * DD, n, m0, i * 32);
  int l = threadIdx.x & 31, h = l >> 4;
  for (int n0 = 0; n0 < 64; n0 += 16) {
    v8f acc = {};
#pragma unroll
    for (int i = 0; i < 8; ++i)
      acc = WMMA_BF16(a[i], frag_b_lds(sP, LDP, n0, i * 32), acc);
    int col = n0 + (l & 15);
    if (!valid || col >= KK) continue;
    float bv = b_phi[col];
#pragma unroll
    for (int r = 0; r < 8; ++r) {
      int m = m0 + r + 8 * h;
      if (full || m < n)
        logits[(size_t)m * KK + col] = acc[r] + bv + docphi[(size_t)xbatch[m] * 64 + col];
    }
  }
}

// phi = softmax_k( BN(logits) ), one thread per token row
__global__ void k_bn_softmax(const float* __restrict__ logits, const float* __restrict__ mean,
                             const float* __restrict__ rstd, const float* __restrict__ g,
                             const float* __restrict__ b, float* __restrict__ out, int n) {
  int i = blockIdx.x * blockDim.x + threadIdx.x;
  if (i >= n) return;
  float v[KK];
  float mx = -1e30f;
#pragma unroll
  for (int k = 0; k < KK; ++k) {
    float y = g[k] * (logits[(size_t)i * KK + k] - mean[k]) * rstd[k] + b[k];
    v[k] = y;
    mx = fmaxf(mx, y);
  }
  float s = 0.f;
#pragma unroll
  for (int k = 0; k < KK; ++k) { v[k] = __expf(v[k] - mx); s += v[k]; }
  float inv = 1.f / s;
#pragma unroll
  for (int k = 0; k < KK; ++k) out[(size_t)i * KK + k] = v[k] * inv;
}

// ---------------- dispatch --------------------------------------------------
static inline size_t align16f(size_t v) { return (v + 15) & ~(size_t)15; }  // 64B

extern "C" void kernel_launch(void* const* d_in, const int* in_sizes, int n_in,
                              void* d_out, int out_size, void* d_ws, size_t ws_size,
                              hipStream_t stream) {
  const int*   idx_x   = (const int*)  d_in[0];
  const float* idx_w   = (const float*)d_in[1];
  const int*   x_batch = (const int*)  d_in[2];
  const int*   e_idx   = (const int*)  d_in[3];
  const float* e_w     = (const float*)d_in[4];
  const float* word_vec= (const float*)d_in[5];
  const float* w_rel   = (const float*)d_in[6];
  const float* b_rel   = (const float*)d_in[7];
  const float* w_root  = (const float*)d_in[8];
  const float* bn1_g   = (const float*)d_in[9];
  const float* bn1_b   = (const float*)d_in[10];
  const float* w_fc1   = (const float*)d_in[11];
  const float* b_fc1   = (const float*)d_in[12];
  const float* w_fc2   = (const float*)d_in[13];
  const float* b_fc2   = (const float*)d_in[14];
  const float* w_mean  = (const float*)d_in[15];
  const float* b_mean  = (const float*)d_in[16];
  const float* w_lv    = (const float*)d_in[17];
  const float* b_lv    = (const float*)d_in[18];
  const float* w_phi   = (const float*)d_in[19];
  const float* b_phi   = (const float*)d_in[20];
  const float* phibn_g = (const float*)d_in[21];
  const float* phibn_b = (const float*)d_in[22];

  const int N = in_sizes[0];
  const int E = in_sizes[4];
  const int ET = E + N;

  float* ws = (float*)d_ws;
  size_t off = 0;
  float* aggr      = ws + off; off += (size_t)N * DD;     // zeroed
  float* enc2      = ws + off; off += (size_t)BB * HH;    // zeroed
  float* gnn_sums  = ws + off; off += DD;                 // zeroed
  float* gnn_sumsq = ws + off; off += DD;                 // zeroed
  float* phi_sums  = ws + off; off += 64;                 // zeroed
  float* phi_sumsq = ws + off; off += 64;                 // zeroed
  size_t zero_count = off;
  float* gnn_mean  = ws + off; off += DD;
  float* gnn_rstd  = ws + off; off += DD;
  float* phi_mean  = ws + off; off += 64;
  float* phi_rstd  = ws + off; off += 64;
  float* gnn       = ws + off; off += (size_t)N * DD;
  float* docphi    = ws + off; off += (size_t)BB * 64;
  float* logits    = ws + off; off += (size_t)N * KK;
  off = align16f(off);
  __bf16* xbf      = (__bf16*)(ws + off); off += (size_t)N * DD / 2;
  __bf16* aggrbf   = (__bf16*)(ws + off); off += (size_t)N * DD / 2;
  __bf16* enc1bf   = (__bf16*)(ws + off); off += (size_t)N * DD;       // [N, 2D]
  __bf16* enc2bf   = (__bf16*)(ws + off); off += (size_t)BB * HH / 2;
  __bf16* w_relbf  = (__bf16*)(ws + off); off += (size_t)DD * DD / 2;
  __bf16* w_rootbf = (__bf16*)(ws + off); off += (size_t)DD * DD / 2;
  __bf16* w_fc1bf  = (__bf16*)(ws + off); off += (size_t)HH * 2 * DD / 2;
  __bf16* w_fc2bf  = (__bf16*)(ws + off); off += (size_t)HH * 2 * DD / 2;
  __bf16* w_meanbf = (__bf16*)(ws + off); off += align16f((size_t)KK * HH / 2);
  __bf16* w_lvbf   = (__bf16*)(ws + off); off += align16f((size_t)KK * HH / 2);
  __bf16* w_phibf  = (__bf16*)(ws + off); off += (size_t)KK * (2 * DD + HH) / 2;

  float* out_mean = (float*)d_out;
  float* out_lv   = out_mean + (size_t)BB * KK;
  float* out_phi  = out_lv   + (size_t)BB * KK;

  const int mblocks = ((N + 15) / 16 + 7) / 8;   // 8 waves (16 rows each) / block

  // 0) zero accumulators; weights -> bf16
  k_zero<<<2048, 256, 0, stream>>>(ws, zero_count);
  k_cast_bf16<<<64, 256, 0, stream>>>(w_rel,  w_relbf,  (size_t)DD * DD);
  k_cast_bf16<<<64, 256, 0, stream>>>(w_root, w_rootbf, (size_t)DD * DD);
  k_cast_bf16<<<256, 256, 0, stream>>>(w_fc1, w_fc1bf,  (size_t)HH * 2 * DD);
  k_cast_bf16<<<256, 256, 0, stream>>>(w_fc2, w_fc2bf,  (size_t)HH * 2 * DD);
  k_cast_bf16<<<64, 256, 0, stream>>>(w_mean, w_meanbf, (size_t)KK * HH);
  k_cast_bf16<<<64, 256, 0, stream>>>(w_lv,   w_lvbf,   (size_t)KK * HH);
  k_cast_bf16<<<128, 256, 0, stream>>>(w_phi, w_phibf,  (size_t)KK * (2 * DD + HH));
  // 1) gather + cast to bf16
  k_gather<<<(N * DD + 255) / 256, 256, 0, stream>>>(idx_x, word_vec, xbf, N * DD);
  // 2) edge + self-loop scatter-add (f32 atomics), then cast aggr -> bf16
  k_edges<<<(ET + 7) / 8, 256, 0, stream>>>(e_idx, e_w, idx_w, xbf, aggr, E, ET);
  k_cast_bf16<<<4096, 256, 0, stream>>>(aggr, aggrbf, (size_t)N * DD);
  // 3) gnn = aggr@w_rel^T + b_rel + x@w_root^T
  k_gemm_gnn<<<mblocks, 256, 0, stream>>>(aggrbf, xbf, w_relbf, w_rootbf, b_rel, gnn, N);
  // 4) BN stats for gnn
  k_colstats<<<(N + 255) / 256, dim3(64, 4), 0, stream>>>(gnn, N, DD, DD, 256,
                                                          gnn_sums, gnn_sumsq);
  k_finalize_stats<<<1, DD, 0, stream>>>(gnn_sums, gnn_sumsq, DD, 1.f / (float)N,
                                         gnn_mean, gnn_rstd);
  // 5) enc1 = [tanh(BN(gnn)), x]  (bf16)
  k_enc1<<<(N * DD + 255) / 256, 256, 0, stream>>>(gnn, xbf, gnn_mean, gnn_rstd,
                                                   bn1_g, bn1_b, enc1bf, N * DD);
  // 6) gated FC + fused segment_sum -> enc2 [B,H]
  k_fc_scatter<<<mblocks, 256, 0, stream>>>(enc1bf, w_fc1bf, b_fc1, w_fc2bf, b_fc2,
                                            x_batch, enc2, N);
  k_cast_bf16<<<256, 256, 0, stream>>>(enc2, enc2bf, (size_t)BB * HH);
  // 7) mean / logvar / per-doc phi term
  k_gemm_small<<<dim3((BB + 15) / 16, 4), 32, 0, stream>>>(enc2bf, HH, BB, w_meanbf, HH,
                                                           KK, b_mean, out_mean, KK, KK, HH);
  k_gemm_small<<<dim3((BB + 15) / 16, 4), 32, 0, stream>>>(enc2bf, HH, BB, w_lvbf, HH,
                                                           KK, b_lv, out_lv, KK, KK, HH);
  k_gemm_small<<<dim3((BB + 15) / 16, 4), 32, 0, stream>>>(enc2bf, HH, BB, w_phibf + 2 * DD,
                                                           2 * DD + HH, KK, nullptr,
                                                           docphi, 64, KK, HH);
  // 8) phi logits = enc1 part + doc part + bias
  k_phi_logits<<<mblocks, 256, 0, stream>>>(enc1bf, w_phibf, b_phi, docphi, x_batch,
                                            logits, N);
  // 9) BN stats + softmax over K
  k_colstats<<<(N + 255) / 256, dim3(64, 4), 0, stream>>>(logits, N, KK, KK, 256,
                                                          phi_sums, phi_sumsq);
  k_finalize_stats<<<1, 64, 0, stream>>>(phi_sums, phi_sumsq, KK, 1.f / (float)N,
                                         phi_mean, phi_rstd);
  k_bn_softmax<<<(N + 127) / 128, 128, 0, stream>>>(logits, phi_mean, phi_rstd,
                                                    phibn_g, phibn_b, out_phi, N);
  (void)n_in; (void)out_size; (void)ws_size;
}
